// GCN_14568529068684
// MI455X (gfx1250) — matrix operand order repaired and checked by reading, not compiled, not run on hardware
//
#include <hip/hip_runtime.h>
#include <math.h>

// ---------------------------------------------------------------------------
// Types for CDNA5 WMMA (wave32, 16x16x32 bf16 -> f32)
// ---------------------------------------------------------------------------
typedef __attribute__((ext_vector_type(16))) __bf16 v16bf;
typedef __attribute__((ext_vector_type(8)))  float  v8f;

__device__ __forceinline__ unsigned pkbf(float a, float b) {
    union { __bf16 h[2]; unsigned u; } t;
    t.h[0] = (__bf16)a; t.h[1] = (__bf16)b;
    return t.u;
}

// Epilogue flags
#define EPI_ADDM 1
#define EPI_BIAS 2
#define EPI_RELU 4

#define BM 128
#define BN 256
#define BK 32

// ---------------------------------------------------------------------------
// WMMA GEMM: C[M,N] = A[M,K](f32, cvt->bf16 on the fly) @ Bt[N,K](bf16)
// compile-time epilogue: += Dadd[M,N], += bias[N], relu
// Block tile 128x256, BK=32, 512 threads (16 waves), double-buffered LDS,
// software-pipelined (global prefetch overlaps WMMA).
// M % 128 == 0, K % 32 == 0. N may be ragged (bounds-checked on store).
// ---------------------------------------------------------------------------
template <int FLAGS>
__global__ __launch_bounds__(512)
void wmma_gemm(const float* __restrict__ A, const __bf16* __restrict__ Bt,
               float* __restrict__ C, const float* __restrict__ Dadd,
               const float* __restrict__ bias,
               int M, int N, int K)
{
    __shared__ __bf16 lA[2][BM * BK];   // [m][k], row stride 32
    __shared__ __bf16 lB[2][BN * BK];   // [n][k], row stride 32

    const int tid  = threadIdx.x;
    const int lane = tid & 31;
    const int wave = tid >> 5;                 // 0..15 -> N sub-block

    const int m0 = blockIdx.y * BM;
    const int n0 = blockIdx.x * BN;

    v8f acc[8] = {};

    // global->reg staging mapping
    const int ar = tid >> 2;                   // 0..127  (A row in tile)
    const int aq = (tid & 3) * 8;              // A col offset (8 f32 per thread)
    const int br = tid >> 1;                   // 0..255  (Bt row in tile)
    const int bh = (tid & 1) * 16;             // Bt col offset (16 bf16 per thread)

    // clamp ragged-N row: garbage only reaches columns >= N, discarded on store
    const int brow = (n0 + br < N) ? (n0 + br) : (N - 1);
    const float*  aptr = A  + (size_t)(m0 + ar) * K + aq;
    const __bf16* bptr = Bt + (size_t)brow * K + bh;

    // fragment mapping (per ISA 16-bit layouts)
    const int a_row = lane & 15;
    const int a_kb  = (lane >> 4) * 8;         // 0 or 8
    const int b_col = wave * 16 + (lane & 15); // 0..255
    const int b_kb  = (lane >> 4) * 8;         // 0 or 8

    uint4 avec;                                // 8 bf16 (A stage)
    uint4 bvec0, bvec1;                        // 16 bf16 (B stage)

    auto g2r = [&](int k0) {
        const float4* ap = (const float4*)(aptr + k0);
        float4 f0 = ap[0];
        float4 f1 = ap[1];
        avec.x = pkbf(f0.x, f0.y); avec.y = pkbf(f0.z, f0.w);
        avec.z = pkbf(f1.x, f1.y); avec.w = pkbf(f1.z, f1.w);
        const uint4* bp = (const uint4*)(bptr + k0);
        bvec0 = bp[0]; bvec1 = bp[1];
    };
    auto r2lds = [&](int buf) {
        *(uint4*)(lA[buf] + ar * BK + aq) = avec;
        uint4* d = (uint4*)(lB[buf] + br * BK + bh);
        d[0] = bvec0; d[1] = bvec1;
    };
    auto compute = [&](int buf) {
        union U { uint4 u[2]; v16bf v; };
        U bf;
        {
            const __bf16* p = lB[buf] + b_col * BK;
            bf.u[0] = *(const uint4*)(p + b_kb); bf.u[1] = *(const uint4*)(p + 16 + b_kb);
        }
        U af[8];
        #pragma unroll
        for (int mi = 0; mi < 8; ++mi) {
            const __bf16* ap = lA[buf] + (mi * 16 + a_row) * BK;
            af[mi].u[0] = *(const uint4*)(ap + a_kb);
            af[mi].u[1] = *(const uint4*)(ap + 16 + a_kb);
        }
        // keep all 16 ds_loads batched before the 8 WMMAs (one dscnt wait,
        // back-to-back XDL issue) instead of per-WMMA reload+wait
        __builtin_amdgcn_sched_barrier(0);
        #pragma unroll
        for (int mi = 0; mi < 8; ++mi) {
            acc[mi] = __builtin_amdgcn_wmma_f32_16x16x32_bf16(
                false, af[mi].v, false, bf.v, (short)0, acc[mi], false, false);
        }
    };

    const int nk = K / BK;
    g2r(0);
    r2lds(0);
    for (int kk = 0; kk < nk - 1; ++kk) {      // peeled: no branches in body
        __syncthreads();
        g2r((kk + 1) * BK);                    // prefetch next tile (global)
        compute(kk & 1);
        r2lds((kk + 1) & 1);                   // stage into the other buffer
    }
    __syncthreads();
    compute((nk - 1) & 1);

    // ---- epilogue + store (C 16x16 f32 layout: VGPR i -> M = (lane<16?i:8+i)) ----
    const int ccol = n0 + wave * 16 + (lane & 15);
    if (ccol < N) {
        const int rbase = m0 + (lane >> 4) * 8;
        const float bv = (FLAGS & EPI_BIAS) ? bias[ccol] : 0.0f;
        #pragma unroll
        for (int mi = 0; mi < 8; ++mi) {
            #pragma unroll
            for (int i = 0; i < 8; ++i) {
                size_t idx = (size_t)(rbase + mi * 16 + i) * N + ccol;
                float v = acc[mi][i] + bv;
                if (FLAGS & EPI_ADDM) v += Dadd[idx];
                if (FLAGS & EPI_RELU) v = fmaxf(v, 0.0f);
                C[idx] = v;
            }
        }
    }
}

// ---------------------------------------------------------------------------
// Transpose + convert: src f32 [R,C] -> dst bf16 [C,R]
// ---------------------------------------------------------------------------
__global__ __launch_bounds__(256)
void transpose_cvt(const float* __restrict__ src, __bf16* __restrict__ dst,
                   int R, int C)
{
    __shared__ float tile[32][33];
    const int cb = blockIdx.x * 32, rb = blockIdx.y * 32;
    const int tx = threadIdx.x & 31, ty = threadIdx.x >> 5;
    for (int i = ty; i < 32; i += 8) {
        int r = rb + i, c = cb + tx;
        tile[i][tx] = (r < R && c < C) ? src[(size_t)r * C + c] : 0.0f;
    }
    __syncthreads();
    for (int i = ty; i < 32; i += 8) {
        int c = cb + i, r = rb + tx;
        if (c < C && r < R) dst[(size_t)c * R + r] = (__bf16)tile[tx][i];
    }
}

// ---------------------------------------------------------------------------
// Per-column batch stats: mean[col], rstd[col] = rsqrt(var+eps). grid.x == N
// ---------------------------------------------------------------------------
__global__ __launch_bounds__(256)
void col_stats(const float* __restrict__ X, int M, int N,
               float* __restrict__ mean, float* __restrict__ rstd)
{
    const int col = blockIdx.x;
    float s = 0.0f, sq = 0.0f;
    for (int r = threadIdx.x; r < M; r += 256) {
        float v = X[(size_t)r * N + col];
        s += v; sq += v * v;
    }
    __shared__ float sh[256], shq[256];
    sh[threadIdx.x] = s; shq[threadIdx.x] = sq;
    __syncthreads();
    for (int o = 128; o > 0; o >>= 1) {
        if (threadIdx.x < o) {
            sh[threadIdx.x]  += sh[threadIdx.x + o];
            shq[threadIdx.x] += shq[threadIdx.x + o];
        }
        __syncthreads();
    }
    if (threadIdx.x == 0) {
        float m = sh[0] / (float)M;
        float v = shq[0] / (float)M - m * m;
        mean[col] = m;
        rstd[col] = rsqrtf(v + 1e-5f);
    }
}

// ---------------------------------------------------------------------------
// BatchNorm apply: Y = g*(X-mean)*rstd + b
// ---------------------------------------------------------------------------
__global__ __launch_bounds__(256)
void bn_apply(const float* __restrict__ X, float* __restrict__ Y,
              const float* __restrict__ mean, const float* __restrict__ rstd,
              const float* __restrict__ g, const float* __restrict__ b,
              int M, int N)
{
    size_t i = (size_t)blockIdx.x * 256 + threadIdx.x;
    if (i < (size_t)M * N) {
        int c = (int)(i % N);
        Y[i] = g[c] * (X[i] - mean[c]) * rstd[c] + b[c];
    }
}

// ---------------------------------------------------------------------------
// Per-graph pooling: 256 graphs x 32 nodes, emb = [h1 h2] (512 cols)
// pooled[g] = [max(512) | mean(512)]   -> [256, 1024]
// ---------------------------------------------------------------------------
__global__ __launch_bounds__(512)
void pool_kernel(const float* __restrict__ h1, const float* __restrict__ h2,
                 float* __restrict__ pooled)
{
    const int g = blockIdx.x;
    const int c = threadIdx.x;                 // 0..511
    const float* src = (c < 256) ? h1 : h2;
    const int col = c & 255;
    const int base = g * 32;
    float mx = -INFINITY, sm = 0.0f;
    #pragma unroll 4
    for (int n = 0; n < 32; ++n) {
        float v = src[(size_t)(base + n) * 256 + col];
        mx = fmaxf(mx, v); sm += v;
    }
    pooled[(size_t)g * 1024 + c]       = mx;
    pooled[(size_t)g * 1024 + 512 + c] = sm * (1.0f / 32.0f);
}

// ---------------------------------------------------------------------------
// Launcher
// ---------------------------------------------------------------------------
extern "C" void kernel_launch(void* const* d_in, const int* in_sizes, int n_in,
                              void* d_out, int out_size, void* d_ws, size_t ws_size,
                              hipStream_t stream)
{
    (void)in_sizes; (void)n_in; (void)out_size; (void)ws_size;
    const int N = 8192, F = 128, H = 256, G = 256;

    const float* x    = (const float*)d_in[0];
    const float* adj  = (const float*)d_in[1];
    /* d_in[2] = seg (sorted, 32 nodes/graph) unused */
    const float* W1   = (const float*)d_in[3];
    const float* Ws1  = (const float*)d_in[4];
    const float* b1   = (const float*)d_in[5];
    const float* g1   = (const float*)d_in[6];
    const float* be1  = (const float*)d_in[7];
    const float* W2   = (const float*)d_in[8];
    const float* Ws2  = (const float*)d_in[9];
    const float* b2   = (const float*)d_in[10];
    const float* g2   = (const float*)d_in[11];
    const float* be2  = (const float*)d_in[12];
    const float* bn0g = (const float*)d_in[13];
    const float* bn0b = (const float*)d_in[14];
    const float* L1W  = (const float*)d_in[15];
    const float* L1b  = (const float*)d_in[16];
    const float* bn1g = (const float*)d_in[17];
    const float* bn1b = (const float*)d_in[18];
    const float* L2W  = (const float*)d_in[19];
    const float* L2b  = (const float*)d_in[20];
    const float* bn2g = (const float*)d_in[21];
    const float* bn2b = (const float*)d_in[22];
    const float* L3W  = (const float*)d_in[23];
    const float* L3b  = (const float*)d_in[24];
    const float* catW = (const float*)d_in[25];
    const float* catb = (const float*)d_in[26];

    // bump allocator over d_ws
    char* wsb = (char*)d_ws;
    size_t off = 0;
    auto alloc = [&](size_t bytes) -> void* {
        void* p = wsb + off;
        off += (bytes + 255) & ~(size_t)255;
        return p;
    };

    float*  T1   = (float*) alloc((size_t)N * H * 4);   // reused as T2
    float*  S1   = (float*) alloc((size_t)N * H * 4);   // reused as S2
    float*  h1   = (float*) alloc((size_t)N * H * 4);
    float*  h2   = (float*) alloc((size_t)N * H * 4);
    __bf16* T1t  = (__bf16*)alloc((size_t)H * N * 2);   // reused as T2t
    __bf16* W1t  = (__bf16*)alloc((size_t)H * F * 2);
    __bf16* Ws1t = (__bf16*)alloc((size_t)H * F * 2);
    __bf16* W2t  = (__bf16*)alloc((size_t)H * H * 2);
    __bf16* Ws2t = (__bf16*)alloc((size_t)H * H * 2);
    float*  pooled = (float*)alloc((size_t)G * 1024 * 4);
    float*  A0   = (float*) alloc((size_t)G * 1024 * 4);
    __bf16* L1Wt = (__bf16*)alloc((size_t)512 * 1024 * 2);
    float*  H1   = (float*) alloc((size_t)G * 512 * 4);
    float*  A1   = (float*) alloc((size_t)G * 512 * 4);
    __bf16* L2Wt = (__bf16*)alloc((size_t)256 * 512 * 2);
    float*  H2   = (float*) alloc((size_t)G * 256 * 4);
    __bf16* L3Wt = (__bf16*)alloc((size_t)128 * 256 * 2);
    __bf16* catWt= (__bf16*)alloc((size_t)12 * 256 * 2);
    float*  mean = (float*) alloc(1024 * 4);
    float*  rstd = (float*) alloc(1024 * 4);

    auto tc = [&](const float* s, __bf16* d, int R, int C) {
        dim3 grid((C + 31) / 32, (R + 31) / 32);
        transpose_cvt<<<grid, 256, 0, stream>>>(s, d, R, C);
    };
    auto gemm = [&](const float* A, const __bf16* Bt, float* C,
                    const float* D, const float* bias,
                    int M_, int N_, int K_, int flags) {
        dim3 grid((N_ + BN - 1) / BN, M_ / BM);
        switch (flags) {
        case 0:
            wmma_gemm<0><<<grid, 512, 0, stream>>>(A, Bt, C, D, bias, M_, N_, K_);
            break;
        case EPI_BIAS:
            wmma_gemm<EPI_BIAS><<<grid, 512, 0, stream>>>(A, Bt, C, D, bias, M_, N_, K_);
            break;
        case EPI_BIAS | EPI_RELU:
            wmma_gemm<EPI_BIAS | EPI_RELU><<<grid, 512, 0, stream>>>(A, Bt, C, D, bias, M_, N_, K_);
            break;
        default:
            wmma_gemm<EPI_ADDM | EPI_BIAS | EPI_RELU><<<grid, 512, 0, stream>>>(A, Bt, C, D, bias, M_, N_, K_);
            break;
        }
    };

    // weight transposes (bf16, [N,K] layout for GEMM B operand)
    tc(W1, W1t, F, H);     tc(Ws1, Ws1t, F, H);
    tc(W2, W2t, H, H);     tc(Ws2, Ws2t, H, H);
    tc(L1W, L1Wt, 1024, 512);
    tc(L2W, L2Wt, 512, 256);
    tc(L3W, L3Wt, 256, 128);
    tc(catW, catWt, 256, 12);

    // ---- GCN layer 1 ----
    gemm(x, W1t,  T1, nullptr, nullptr, N, H, F, 0);                  // T1 = x@W1
    gemm(x, Ws1t, S1, nullptr, nullptr, N, H, F, 0);                  // S1 = x@Ws1
    tc(T1, T1t, N, H);
    gemm(adj, T1t, h1, S1, b1, N, H, N, EPI_ADDM | EPI_BIAS | EPI_RELU); // relu(adj@T1+b1+S1)
    col_stats<<<H, 256, 0, stream>>>(h1, N, H, mean, rstd);
    bn_apply<<<(N * H + 255) / 256, 256, 0, stream>>>(h1, h1, mean, rstd, g1, be1, N, H);

    // ---- GCN layer 2 ----
    gemm(h1, W2t,  T1, nullptr, nullptr, N, H, H, 0);                 // T2 (reuse T1)
    gemm(h1, Ws2t, S1, nullptr, nullptr, N, H, H, 0);                 // S2 (reuse S1)
    tc(T1, T1t, N, H);
    gemm(adj, T1t, h2, S1, b2, N, H, N, EPI_ADDM | EPI_BIAS | EPI_RELU);
    col_stats<<<H, 256, 0, stream>>>(h2, N, H, mean, rstd);
    bn_apply<<<(N * H + 255) / 256, 256, 0, stream>>>(h2, h2, mean, rstd, g2, be2, N, H);

    // ---- pooling ----
    pool_kernel<<<G, 512, 0, stream>>>(h1, h2, pooled);               // [256,1024]

    // ---- MLP head ----
    col_stats<<<1024, 256, 0, stream>>>(pooled, G, 1024, mean, rstd);
    bn_apply<<<(G * 1024 + 255) / 256, 256, 0, stream>>>(pooled, A0, mean, rstd, bn0g, bn0b, G, 1024);
    gemm(A0, L1Wt, H1, nullptr, L1b, G, 512, 1024, EPI_BIAS | EPI_RELU);

    col_stats<<<512, 256, 0, stream>>>(H1, G, 512, mean, rstd);
    bn_apply<<<(G * 512 + 255) / 256, 256, 0, stream>>>(H1, A1, mean, rstd, bn1g, bn1b, G, 512);
    gemm(A1, L2Wt, H2, nullptr, L2b, G, 256, 512, EPI_BIAS | EPI_RELU);

    // outputs: [out 256x128 | out_class 256x12 | fingerprint 256x256]
    float* out       = (float*)d_out;
    float* out_class = out + 256 * 128;
    float* fp        = out_class + 256 * 12;

    col_stats<<<256, 256, 0, stream>>>(H2, G, 256, mean, rstd);
    bn_apply<<<(G * 256 + 255) / 256, 256, 0, stream>>>(H2, fp, mean, rstd, bn2g, bn2b, G, 256);

    gemm(fp, L3Wt,  out,       nullptr, L3b,  G, 128, 256, EPI_BIAS);
    gemm(fp, catWt, out_class, nullptr, catb, G, 12,  256, EPI_BIAS);
}
